// DecoderModule_3272765079787
// MI455X (gfx1250) — compile-verified, matrix operand order checked by reading, not run on hardware
//
#include <hip/hip_runtime.h>
#include <hip/hip_bf16.h>

// ---------------- problem constants ----------------
#define B_    64
#define T_    128
#define S_    512
#define E_    256
#define H_    512
#define H2_   1024        // 2*H
#define GRU_IN_  1280     // E + 2H
#define PRE_IN_  1792     // E + H + 2H
#define G3_   1536        // 3*H
#define QGH_  2048        // H (query) + 3H (gru hidden gates)

typedef __bf16 bf16;
typedef __attribute__((ext_vector_type(16))) __bf16 v16bf;
typedef __attribute__((ext_vector_type(8)))  __bf16 v8bf;
typedef __attribute__((ext_vector_type(8)))  float  v8f;

// ---------------- WMMA fragment helpers ----------------
// CDNA5 16-bit A-matrix 16x32 layout (CDNA5 ISA):
//   lane L (row = L%16): L<16 holds K={0..7,16..23}, L>=16 holds K={8..15,24..31}
// B-matrix (K x N) mirrors this with column n = L%16 when sourced from a
// row-major [N,K] weight matrix (i.e. same load pattern as A).
__device__ inline v16bf load_frag_bf16(const bf16* p, int ld) {
    int lane = threadIdx.x & 31;
    const bf16* r = p + (size_t)(lane & 15) * ld + ((lane < 16) ? 0 : 8);
    v8bf lo = *(const v8bf*)(r);
    v8bf hi = *(const v8bf*)(r + 16);
    v16bf f;
#pragma unroll
    for (int i = 0; i < 8; ++i) { f[i] = lo[i]; f[i + 8] = hi[i]; }
    return f;
}

// Generic C[MxN] = A[MxK] * W[NxK]^T (+bias).
// One wave computes a 16x64 C tile (4 WMMA accumulators). The K loop is
// unrolled x2 with ping-pong fragment buffers: loads for tile k+1 overlap
// the WMMAs of tile k, with no register-rotation copies.
// Requirements: N % 64 == 0, M % 16 == 0, K % 64 == 0 (even # of K-tiles).
template<bool STORE_F32, bool STORE_BF16>
__global__ __launch_bounds__(256)
void gemm_wmma(const bf16* __restrict__ A,  int lda,
               const bf16* __restrict__ W,  int ldw,
               const float* __restrict__ bias,
               float* __restrict__ C,  int ldc,
               bf16*  __restrict__ Cb, int ldcb,
               int M, int N, int K)
{
    constexpr int NB = 4;                     // 4 x 16 = 64 N-columns per wave
    int nBlk  = N >> 6;
    int wid   = (int)((blockIdx.x * blockDim.x + threadIdx.x) >> 5);
    int total = (M >> 4) * nBlk;
    if (wid >= total) return;
    int tm = wid / nBlk, tb = wid - tm * nBlk;
    int m0 = tm << 4, n0 = tb << 6;

    const bf16* pa = A + (size_t)m0 * lda;
    const bf16* pw = W + (size_t)n0 * ldw;

    v8f acc[NB] = {};
    v16bf a0, a1, b0[NB], b1[NB];

    // prologue: tile 0 -> ping
    a0 = load_frag_bf16(pa, lda);
#pragma unroll
    for (int j = 0; j < NB; ++j)
        b0[j] = load_frag_bf16(pw + (size_t)(16 * j) * ldw, ldw);

    int K32 = K >> 5;                         // even by contract
    for (int kt = 0; kt < K32 - 2; kt += 2) {
        int k1 = (kt + 1) << 5, k2 = (kt + 2) << 5;
        // load tile kt+1 -> pong
        a1 = load_frag_bf16(pa + k1, lda);
#pragma unroll
        for (int j = 0; j < NB; ++j)
            b1[j] = load_frag_bf16(pw + (size_t)(16 * j) * ldw + k1, ldw);
        // drain tile kt (ping)
#pragma unroll
        for (int j = 0; j < NB; ++j)
            acc[j] = __builtin_amdgcn_wmma_f32_16x16x32_bf16(
                         false, a0, false, b0[j], (short)0, acc[j], false, false);
        // load tile kt+2 -> ping
        a0 = load_frag_bf16(pa + k2, lda);
#pragma unroll
        for (int j = 0; j < NB; ++j)
            b0[j] = load_frag_bf16(pw + (size_t)(16 * j) * ldw + k2, ldw);
        // drain tile kt+1 (pong)
#pragma unroll
        for (int j = 0; j < NB; ++j)
            acc[j] = __builtin_amdgcn_wmma_f32_16x16x32_bf16(
                         false, a1, false, b1[j], (short)0, acc[j], false, false);
    }
    // epilogue: tiles K32-2 (in ping) and K32-1
    {
        int kl = (K32 - 1) << 5;
        a1 = load_frag_bf16(pa + kl, lda);
#pragma unroll
        for (int j = 0; j < NB; ++j)
            b1[j] = load_frag_bf16(pw + (size_t)(16 * j) * ldw + kl, ldw);
#pragma unroll
        for (int j = 0; j < NB; ++j)
            acc[j] = __builtin_amdgcn_wmma_f32_16x16x32_bf16(
                         false, a0, false, b0[j], (short)0, acc[j], false, false);
#pragma unroll
        for (int j = 0; j < NB; ++j)
            acc[j] = __builtin_amdgcn_wmma_f32_16x16x32_bf16(
                         false, a1, false, b1[j], (short)0, acc[j], false, false);
    }

    int lane = threadIdx.x & 31;
    int nl   = lane & 15;
    int mb   = m0 + ((lane < 16) ? 0 : 8);
#pragma unroll
    for (int j = 0; j < NB; ++j) {
        int n = n0 + 16 * j + nl;
        float bv = bias ? bias[n] : 0.0f;
#pragma unroll
        for (int r = 0; r < 8; ++r) {
            float v = acc[j][r] + bv;
            if (STORE_F32)  C [(size_t)(mb + r) * ldc  + n] = v;
            if (STORE_BF16) Cb[(size_t)(mb + r) * ldcb + n] = (bf16)v;
        }
    }
}

// ---------------- small utility kernels ----------------
__global__ void f32_to_bf16_kernel(const float* __restrict__ in,
                                   bf16* __restrict__ out, int n) {
    int i = blockIdx.x * blockDim.x + threadIdx.x;
    if (i < n) out[i] = (bf16)in[i];
}

// biascat = [zeros(H) | b_hh(3H)]  for the fused q/gh GEMM
__global__ void build_biascat_kernel(const float* __restrict__ b_hh,
                                     float* __restrict__ biascat) {
    int i = blockIdx.x * blockDim.x + threadIdx.x;   // QGH_
    biascat[i] = (i < H_) ? 0.0f : b_hh[i - H_];
}

// h0 = tanh(enc_final[0] @ W_bridge^T + b_bridge)   [B,H]
__global__ void bridge_kernel(const float* __restrict__ enc_final,
                              const float* __restrict__ Wb,
                              const float* __restrict__ bb,
                              float* __restrict__ h, bf16* __restrict__ hb) {
    int idx = blockIdx.x * blockDim.x + threadIdx.x;   // B*H = 32768
    int b = idx >> 9, j = idx & (H_ - 1);
    const float* x = enc_final + (size_t)b * H2_;
    const float* w = Wb        + (size_t)j * H2_;
    float acc = bb[j];
    for (int k = 0; k < H2_; ++k) acc += x[k] * w[k];
    float v = tanhf(acc);
    h[idx]  = v;
    hb[idx] = (bf16)v;
}

// Per-step attention: energy -> masked softmax -> context; builds xcat/precat.
// One workgroup (256 thr = 8 waves) per batch row. proj_key and enc_hidden
// are bf16 (L2-resident working set ~99MB on the 192MB L2).
__global__ __launch_bounds__(256)
void attn_kernel(const float* __restrict__ qgh,        // [B, QGH_], q in cols 0..H-1
                 const bf16*  __restrict__ pk,
                 const float* __restrict__ we_g,
                 const int*   __restrict__ mask,
                 const bf16*  __restrict__ enc_b,      // [B,S,2H] bf16
                 const bf16*  __restrict__ trg_b,
                 bf16* __restrict__ xcat,
                 bf16* __restrict__ precat,
                 int t)
{
    __shared__ float qs[H_];
    __shared__ float wes[H_];
    __shared__ float e[S_];
    __shared__ float red[8];

    int b    = blockIdx.x;
    int tid  = threadIdx.x;
    int lane = tid & 31, wv = tid >> 5;

    for (int i = tid; i < H_; i += 256) {
        qs[i]  = qgh[(size_t)b * QGH_ + i];
        wes[i] = we_g[i];
    }
    __syncthreads();

    // energy: e[s] = sum_h tanh(q[h] + pk[b,s,h]) * we[h]
    const bf16* pkb  = pk   + (size_t)b * S_ * H_;
    const int*  mrow = mask + (size_t)b * S_;
    for (int s = wv * 64; s < wv * 64 + 64; ++s) {
        float p = 0.f;
        const bf16* row = pkb + (size_t)s * H_;
#pragma unroll 4
        for (int k = lane; k < H_; k += 32)
            p += tanhf(qs[k] + (float)row[k]) * wes[k];
        for (int off = 16; off; off >>= 1) p += __shfl_xor(p, off, 32);
        if (lane == 0) e[s] = (mrow[s] != 0) ? p : -1e9f;
    }
    __syncthreads();

    // softmax over S=512 (each thread owns e[tid], e[tid+256])
    float m2 = fmaxf(e[tid], e[tid + 256]);
    for (int off = 16; off; off >>= 1) m2 = fmaxf(m2, __shfl_xor(m2, off, 32));
    if (lane == 0) red[wv] = m2;
    __syncthreads();
    float mx = red[0];
#pragma unroll
    for (int i = 1; i < 8; ++i) mx = fmaxf(mx, red[i]);
    __syncthreads();

    float e0 = __expf(e[tid] - mx), e1 = __expf(e[tid + 256] - mx);
    float s2 = e0 + e1;
    for (int off = 16; off; off >>= 1) s2 += __shfl_xor(s2, off, 32);
    if (lane == 0) red[wv] = s2;
    __syncthreads();
    float sum = 0.f;
#pragma unroll
    for (int i = 0; i < 8; ++i) sum += red[i];
    float inv = 1.0f / sum;
    e[tid] = e0 * inv; e[tid + 256] = e1 * inv;
    __syncthreads();

    // context[b,d] = sum_s alpha[s] * enc_hidden[b,s,d]   (d < 2H)
    const bf16* encb = enc_b + (size_t)b * S_ * H2_;
    for (int d = tid; d < H2_; d += 256) {
        float accv = 0.f;
        for (int s = 0; s < S_; ++s)
            accv += e[s] * (float)encb[(size_t)s * H2_ + d];
        bf16 cv = (bf16)accv;
        xcat  [(size_t)b * GRU_IN_ + E_ + d]      = cv;   // [x_t | ctx]
        precat[(size_t)b * PRE_IN_ + E_ + H_ + d] = cv;   // [x_t | h_new | ctx]
    }
    // copy x_t (bf16, precomputed)
    const bf16* xt = trg_b + ((size_t)b * T_ + t) * E_;
    for (int i = tid; i < E_; i += 256) {
        bf16 v = xt[i];
        xcat  [(size_t)b * GRU_IN_ + i] = v;
        precat[(size_t)b * PRE_IN_ + i] = v;
    }
}

// GRU gate elementwise math + state update + outputs
__global__ void gates_kernel(const float* __restrict__ gx,     // [B, 3H]
                             const float* __restrict__ qgh,    // gh at cols H..H+3H-1
                             float* __restrict__ h, bf16* __restrict__ hb,
                             bf16* __restrict__ precat,
                             float* __restrict__ states_out, int t)
{
    int idx = blockIdx.x * blockDim.x + threadIdx.x;   // B*H
    int b = idx >> 9, j = idx & (H_ - 1);
    const float* gxr = gx  + (size_t)b * G3_;
    const float* ghr = qgh + (size_t)b * QGH_ + H_;
    float r = 1.f / (1.f + __expf(-(gxr[j]          + ghr[j])));
    float z = 1.f / (1.f + __expf(-(gxr[H_ + j]     + ghr[H_ + j])));
    float n = tanhf(gxr[2 * H_ + j] + r * ghr[2 * H_ + j]);
    float hn = (1.f - z) * n + z * h[idx];
    h[idx]  = hn;
    hb[idx] = (bf16)hn;
    precat[(size_t)b * PRE_IN_ + E_ + j] = (bf16)hn;
    states_out[((size_t)b * T_ + t) * H_ + j] = hn;
}

__global__ void copy_f32_kernel(const float* __restrict__ in,
                                float* __restrict__ out, int n) {
    int i = blockIdx.x * blockDim.x + threadIdx.x;
    if (i < n) out[i] = in[i];
}

// ---------------- host driver ----------------
extern "C" void kernel_launch(void* const* d_in, const int* in_sizes, int n_in,
                              void* d_out, int out_size, void* d_ws, size_t ws_size,
                              hipStream_t stream) {
    const float* trg_embed = (const float*)d_in[0];
    const float* enc_hid   = (const float*)d_in[1];
    const float* enc_final = (const float*)d_in[2];
    const int*   data_mask = (const int*)  d_in[3];
    const float* W_key     = (const float*)d_in[4];
    const float* W_query   = (const float*)d_in[5];
    const float* w_energy  = (const float*)d_in[6];
    const float* W_bridge  = (const float*)d_in[7];
    const float* b_bridge  = (const float*)d_in[8];
    const float* W_ih      = (const float*)d_in[9];
    const float* W_hh      = (const float*)d_in[10];
    const float* b_ih      = (const float*)d_in[11];
    const float* b_hh      = (const float*)d_in[12];
    const float* W_pre     = (const float*)d_in[13];

    // workspace layout (256B aligned)
    char* ws = (char*)d_ws;
    size_t off = 0;
    auto alloc = [&](size_t bytes) -> char* {
        char* p = ws + off;
        off = (off + bytes + 255) & ~(size_t)255;
        return p;
    };
    bf16*  enc_b  = (bf16*) alloc((size_t)B_ * S_ * H2_ * 2);      // 64 MB
    bf16*  pk     = (bf16*) alloc((size_t)B_ * S_ * H_ * 2);       // 32 MB
    bf16*  trg_b  = (bf16*) alloc((size_t)B_ * T_ * E_ * 2);       //  4 MB
    bf16*  Wk     = (bf16*) alloc((size_t)H_ * H2_ * 2);
    bf16*  Wqh    = (bf16*) alloc((size_t)QGH_ * H_ * 2);          // [W_query; W_hh]
    bf16*  Wih    = (bf16*) alloc((size_t)G3_ * GRU_IN_ * 2);
    bf16*  Wpre   = (bf16*) alloc((size_t)H_ * PRE_IN_ * 2);
    float* biascat= (float*)alloc((size_t)QGH_ * 4);
    float* h      = (float*)alloc((size_t)B_ * H_ * 4);
    bf16*  hb     = (bf16*) alloc((size_t)B_ * H_ * 2);
    float* qgh    = (float*)alloc((size_t)B_ * QGH_ * 4);          // [q | gh]
    bf16*  xcat   = (bf16*) alloc((size_t)B_ * GRU_IN_ * 2);
    bf16*  precat = (bf16*) alloc((size_t)B_ * PRE_IN_ * 2);
    float* gx     = (float*)alloc((size_t)B_ * G3_ * 4);
    (void)ws_size; (void)n_in; (void)in_sizes; (void)out_size;

    // output layout: states [B,T,H] | hidden [1,B,H] | pre [B,T,H]
    float* out_states = (float*)d_out;
    float* out_hidden = out_states + (size_t)B_ * T_ * H_;
    float* out_pre    = out_hidden + (size_t)B_ * H_;

    auto cvt = [&](const float* src, bf16* dst, int n) {
        f32_to_bf16_kernel<<<(n + 255) / 256, 256, 0, stream>>>(src, dst, n);
    };
    cvt(enc_hid,   enc_b, B_ * S_ * H2_);
    cvt(trg_embed, trg_b, B_ * T_ * E_);
    cvt(W_key,     Wk,    H_ * H2_);
    cvt(W_query,   Wqh,               H_ * H_);       // rows 0..H-1
    cvt(W_hh,      Wqh + (size_t)H_ * H_, G3_ * H_);  // rows H..QGH-1
    cvt(W_ih,      Wih,   G3_ * GRU_IN_);
    cvt(W_pre,     Wpre,  H_ * PRE_IN_);
    build_biascat_kernel<<<QGH_ / 256, 256, 0, stream>>>(b_hh, biascat);

    // h0 = tanh(enc_final @ W_bridge^T + b)
    bridge_kernel<<<(B_ * H_) / 256, 256, 0, stream>>>(enc_final, W_bridge, b_bridge, h, hb);

    auto gemm_f32 = [&](const bf16* A, int lda, const bf16* W, int ldw,
                        const float* bias, float* C, int ldc,
                        int M, int N, int K) {
        int total = (M >> 4) * (N >> 6);
        gemm_wmma<true, false><<<(total + 7) / 8, 256, 0, stream>>>(
            A, lda, W, ldw, bias, C, ldc, nullptr, 0, M, N, K);
    };

    // proj_key = enc_hidden @ W_key^T  (M=32768, N=512, K=1024) -> bf16
    {
        int M = B_ * S_;
        int total = (M >> 4) * (H_ >> 6);
        gemm_wmma<false, true><<<(total + 7) / 8, 256, 0, stream>>>(
            enc_b, H2_, Wk, H2_, nullptr, nullptr, 0, pk, H_, M, H_, H2_);
    }

    for (int t = 0; t < T_; ++t) {
        // fused [q | gh] = h @ [W_query; W_hh]^T + [0 | b_hh]  (64x2048, K=512)
        gemm_f32(hb, H_, Wqh, H_, biascat, qgh, QGH_, B_, QGH_, H_);
        // attention: energy / softmax / context, build xcat & precat
        attn_kernel<<<B_, 256, 0, stream>>>(qgh, pk, w_energy, data_mask,
                                            enc_b, trg_b, xcat, precat, t);
        // gx = xcat @ W_ih^T + b_ih   (64x1536, K=1280)
        gemm_f32(xcat, GRU_IN_, Wih, GRU_IN_, b_ih, gx, G3_, B_, G3_, GRU_IN_);
        // gate math, update h / hb / precat / states output
        gates_kernel<<<(B_ * H_) / 256, 256, 0, stream>>>(gx, qgh, h, hb, precat,
                                                          out_states, t);
        // pre = precat @ W_pre^T  -> out_pre[:, t, :]   (64x512, K=1792)
        gemm_f32(precat, PRE_IN_, Wpre, PRE_IN_, nullptr,
                 out_pre + (size_t)t * H_, T_ * H_, B_, H_, PRE_IN_);
    }
    // hidden = h_last
    copy_f32_kernel<<<(B_ * H_) / 256, 256, 0, stream>>>(h, out_hidden, B_ * H_);
}
